// MoeLlamaMLP_94489280662
// MI455X (gfx1250) — compile-verified, matrix operand order checked by reading, not run on hardware
//
#include <hip/hip_runtime.h>
#include <hip/hip_bf16.h>
#include <math.h>

// ---------------------------------------------------------------------------
// MoE SwiGLU MLP for MI455X (gfx1250), bf16 WMMA path.
//   H=1024, I=4096, E=8, K=2, T=4096 tokens, fp32 in/out.
// Weights/activations pre-converted to bf16 once; routed GEMMs run pure
// v_wmma_f32_16x16x32_bf16 with async double-buffered LDS A-staging.
// ---------------------------------------------------------------------------

typedef __attribute__((ext_vector_type(16))) __bf16 v16bf;
typedef __attribute__((ext_vector_type(8)))  float  v8f;
typedef __attribute__((address_space(3))) unsigned short lds_ushort;

constexpr int Ecnt = 8;          // experts
constexpr int Tn   = 4096;       // tokens (B*S)
constexpr int Hn   = 1024;       // hidden
constexpr int In   = 4096;       // intermediate
constexpr int MAXS = 8192;       // per-expert slot capacity (2T worst case)
constexpr int BM   = 64;         // rows per GEMM job (expert-local)
constexpr int MAX_TILES = (2 * Tn) / BM + Ecnt;   // 136 worst-case jobs
constexpr int LDS_ROW   = 72;                     // padded row, shorts
constexpr int LDS_BUFB  = BM * LDS_ROW * 2;       // bytes per A buffer

union ABuf { uint4 q[2]; v16bf v; };

// pack two fp32 into packed bf16 pair (round-half-up) with one v_perm_b32
__device__ __forceinline__ unsigned pack2bf(float a, float b) {
  unsigned ua = __builtin_bit_cast(unsigned, a) + 0x8000u;
  unsigned ub = __builtin_bit_cast(unsigned, b) + 0x8000u;
  return __builtin_amdgcn_perm(ub, ua, 0x07060302u);  // {b.hi16, a.hi16}
}

// ---------------------------------------------------------------------------
// Streaming fp32 -> bf16 conversion (8 elements / thread / iter)
// ---------------------------------------------------------------------------
__global__ void __launch_bounds__(256) cvt_bf16_kernel(
    const float* __restrict__ src, unsigned short* __restrict__ dst, long long n)
{
  long long i = ((long long)blockIdx.x * blockDim.x + threadIdx.x) * 8;
  long long stride = (long long)gridDim.x * blockDim.x * 8;
  for (; i < n; i += stride) {
    float4 f0 = *(const float4*)(src + i);
    float4 f1 = *(const float4*)(src + i + 4);
    uint4 o;
    o.x = pack2bf(f0.x, f0.y);
    o.y = pack2bf(f0.z, f0.w);
    o.z = pack2bf(f1.x, f1.y);
    o.w = pack2bf(f1.z, f1.w);
    *(uint4*)(dst + i) = o;
  }
}

// ---------------------------------------------------------------------------
// Router: one wave32 per token. logits = x[t] . rw[e] (fp32), top-2, softmax,
// scatter (token, prob) into per-expert lists.
// ---------------------------------------------------------------------------
__global__ void __launch_bounds__(256) router_topk_kernel(
    const float* __restrict__ act, const float* __restrict__ rw, int din,
    int* __restrict__ counts, int* __restrict__ toks, float* __restrict__ probs)
{
  int gwave = (blockIdx.x * blockDim.x + threadIdx.x) >> 5;
  int lane  = threadIdx.x & 31;
  if (gwave >= Tn) return;

  const float* xr = act + (size_t)gwave * din;
  float logit[Ecnt];
#pragma unroll
  for (int e = 0; e < Ecnt; ++e) {
    const float* wr = rw + (size_t)e * din;
    float acc = 0.f;
    for (int d = lane; d < din; d += 32) acc = fmaf(xr[d], wr[d], acc);
#pragma unroll
    for (int off = 16; off > 0; off >>= 1) acc += __shfl_xor(acc, off, 32);
    logit[e] = acc;
  }

  if (lane == 0) {
    int i0 = 0;
#pragma unroll
    for (int e = 1; e < Ecnt; ++e) if (logit[e] > logit[i0]) i0 = e;
    int i1 = -1; float v1 = 0.f;
#pragma unroll
    for (int e = 0; e < Ecnt; ++e) {
      if (e == i0) continue;
      if (i1 < 0 || logit[e] > v1) { i1 = e; v1 = logit[e]; }
    }
    float ex = expf(v1 - logit[i0]);          // v1 <= v0
    float p0 = 1.f / (1.f + ex);
    float p1 = ex * p0;
    int s0 = atomicAdd(&counts[i0], 1);
    toks[i0 * MAXS + s0]  = gwave;
    probs[i0 * MAXS + s0] = p0;
    int s1 = atomicAdd(&counts[i1], 1);
    toks[i1 * MAXS + s1]  = gwave;
    probs[i1 * MAXS + s1] = p1;
  }
}

// ---------------------------------------------------------------------------
// Build flat job list of BM-row tiles: map[t] = (expert<<16)|local_tile, -1 idle.
// ---------------------------------------------------------------------------
__global__ void tilemap_kernel(const int* __restrict__ counts, int* __restrict__ map)
{
  if (blockIdx.x == 0 && threadIdx.x == 0) {
    int t = 0;
    for (int e = 0; e < Ecnt; ++e) {
      int nt = (counts[e] + BM - 1) / BM;
      for (int j = 0; j < nt; ++j) map[t++] = (e << 16) | j;
    }
    while (t < MAX_TILES) map[t++] = -1;
  }
}

// ---------------------------------------------------------------------------
// Routed GEMM: 256 threads = 8 waves. Block tile M=64 x N=128, bf16 in.
// A rows gathered with global_load_async_to_lds_b128 (double-buffered);
// fragments batched into distinct registers so loads clause under one wait
// and the 4 WMMAs issue back-to-back. Epilogue scales by routing prob and
// scatters with global_atomic_add_f32.
// ---------------------------------------------------------------------------
__global__ void __launch_bounds__(256) routed_gemm_wmma(
    const unsigned short* __restrict__ abf, const unsigned short* __restrict__ wbf,
    float* __restrict__ out,
    const int* __restrict__ map, const int* __restrict__ counts,
    const int* __restrict__ toks, const float* __restrict__ probs,
    int din, int dout)
{
  __shared__ unsigned short ldsA[2][BM][LDS_ROW];   // 2 x 64 x 64 bf16 (+pad)
  __shared__ int   s_tok[BM];
  __shared__ float s_p[BM];

  int ment = map[blockIdx.x];
  if (ment < 0) return;
  int e   = ment >> 16;
  int jt  = ment & 0xffff;
  int cnt = counts[e];
  int tid = threadIdx.x;

  if (tid < BM) {
    int j = jt * BM + tid;
    bool ok = j < cnt;
    s_tok[tid] = ok ? toks[e * MAXS + j]  : -1;
    s_p[tid]   = ok ? probs[e * MAXS + j] : 0.f;
  }
  // pre-zero both LDS buffers (pad rows stay zero; valid rows overwritten)
  for (int idx = tid; idx < 2 * BM * LDS_ROW / 2; idx += 256)
    ((unsigned*)ldsA)[idx] = 0u;
  __syncthreads();

  int wave = tid >> 5;
  int lane = tid & 31;
  int n    = blockIdx.y * 128 + wave * 16 + (lane & 15);
  int kb   = (lane >> 4) * 8;      // K-half base per CDNA5 16-bit layout
  int mrow = lane & 15;
  const unsigned short* wrow = wbf + ((size_t)e * dout + n) * din;

  // ---- hoisted async-gather state: this thread moves two b128 segments ----
  int row0 = tid >> 3,          seg0 = tid & 7;
  int row1 = (tid + 256) >> 3;  // seg identical (tid+256 & 7 == seg0)
  int tk0 = s_tok[row0];
  int tk1 = s_tok[row1];
  const unsigned short* gp0 = abf + (size_t)(tk0 < 0 ? 0 : tk0) * din + seg0 * 8;
  const unsigned short* gp1 = abf + (size_t)(tk1 < 0 ? 0 : tk1) * din + seg0 * 8;
  unsigned lb0 = (unsigned)(size_t)(lds_ushort*)&ldsA[0][row0][seg0 * 8];
  unsigned lb1 = (unsigned)(size_t)(lds_ushort*)&ldsA[0][row1][seg0 * 8];

  auto issue_slab = [&](int k0, int pb) {
    unsigned boff = (unsigned)pb * LDS_BUFB;
    if (tk0 >= 0)
      asm volatile("global_load_async_to_lds_b128 %0, %1, off"
                   :: "v"(lb0 + boff), "v"(gp0 + k0) : "memory");
    if (tk1 >= 0)
      asm volatile("global_load_async_to_lds_b128 %0, %1, off"
                   :: "v"(lb1 + boff), "v"(gp1 + k0) : "memory");
  };

  v8f c[4];
#pragma unroll
  for (int mt = 0; mt < 4; ++mt) c[mt] = (v8f){0.f,0.f,0.f,0.f,0.f,0.f,0.f,0.f};

  issue_slab(0, 0);
  int pb = 0;
  for (int k0 = 0; k0 < din; k0 += 64) {
    asm volatile("s_wait_asynccnt 0" ::: "memory");
    __syncthreads();                     // slab pb fully in LDS, all waves
    if (k0 + 64 < din) {
      __builtin_prefetch(wrow + k0 + 64, 0, 1);   // global_prefetch_b8
      issue_slab(k0 + 64, pb ^ 1);
    }
    const unsigned short* wp0 = wrow + k0 + kb;

    // batch both B fragments (4x global_load_b128)
    ABuf bb0, bb1;
    bb0.q[0] = *(const uint4*)(wp0);
    bb0.q[1] = *(const uint4*)(wp0 + 16);
    bb1.q[0] = *(const uint4*)(wp0 + 32);
    bb1.q[1] = *(const uint4*)(wp0 + 48);

#pragma unroll
    for (int ch = 0; ch < 2; ++ch) {
      int kc = ch * 32;
      // batch all 4 A fragments (8x ds_load_b128) into distinct registers
      ABuf aa[4];
#pragma unroll
      for (int mt = 0; mt < 4; ++mt) {
        const unsigned short* ar = &ldsA[pb][mt * 16 + mrow][kc + kb];
        aa[mt].q[0] = *(const uint4*)(ar);
        aa[mt].q[1] = *(const uint4*)(ar + 16);
      }
      const v16bf& bv = ch ? bb1.v : bb0.v;
#pragma unroll
      for (int mt = 0; mt < 4; ++mt)
        c[mt] = __builtin_amdgcn_wmma_f32_16x16x32_bf16(
                    false, aa[mt].v, false, bv, (short)0, c[mt], false, false);
    }
    pb ^= 1;
  }

  // C layout: VGPR v, lane l -> M = v + 8*(l>=16), N = l&15
  int mb = (lane >> 4) * 8;
#pragma unroll
  for (int mt = 0; mt < 4; ++mt) {
#pragma unroll
    for (int v = 0; v < 8; ++v) {
      int m  = mt * 16 + mb + v;
      int tk = s_tok[m];
      if (tk >= 0)
        atomicAdd(&out[(size_t)tk * dout + n], c[mt][v] * s_p[m]);
    }
  }
}

// ---------------------------------------------------------------------------
// SwiGLU: inter = silu(g) * u  -> g (fp32, for down-router) + ibf (bf16)
// ---------------------------------------------------------------------------
__global__ void __launch_bounds__(256) swiglu_kernel(
    float4* __restrict__ g, const float4* __restrict__ u,
    uint2* __restrict__ ibf, int nvec)
{
  int idx    = blockIdx.x * blockDim.x + threadIdx.x;
  int stride = gridDim.x * blockDim.x;
  for (int i = idx; i < nvec; i += stride) {
    float4 gv = g[i];
    float4 uv = u[i];
    float4 r;
    r.x = gv.x / (1.f + expf(-gv.x)) * uv.x;
    r.y = gv.y / (1.f + expf(-gv.y)) * uv.y;
    r.z = gv.z / (1.f + expf(-gv.z)) * uv.z;
    r.w = gv.w / (1.f + expf(-gv.w)) * uv.w;
    g[i] = r;
    uint2 o;
    o.x = pack2bf(r.x, r.y);
    o.y = pack2bf(r.z, r.w);
    ibf[i] = o;
  }
}

// ---------------------------------------------------------------------------
extern "C" void kernel_launch(void* const* d_in, const int* in_sizes, int n_in,
                              void* d_out, int out_size, void* d_ws, size_t ws_size,
                              hipStream_t stream)
{
  (void)in_sizes; (void)n_in; (void)out_size; (void)ws_size;

  const float* x      = (const float*)d_in[0];
  const float* r_gate = (const float*)d_in[1];
  const float* w_gate = (const float*)d_in[2];
  const float* r_up   = (const float*)d_in[3];
  const float* w_up   = (const float*)d_in[4];
  const float* r_down = (const float*)d_in[5];
  const float* w_down = (const float*)d_in[6];
  float* out = (float*)d_out;

  // -------- workspace layout --------
  constexpr size_t MB = 1ull << 20;
  char* ws = (char*)d_ws;
  int* counts   = (int*)ws;                     // 3 x 8 ints
  int* counts_g = counts;
  int* counts_u = counts + 8;
  int* counts_d = counts + 16;
  int* maps  = (int*)(ws + 256);                // 3 x MAX_TILES ints
  int* map_g = maps;
  int* map_u = maps + MAX_TILES;
  int* map_d = maps + 2 * MAX_TILES;
  int* toks  = (int*)(ws + 8192);               // 3 x E x MAXS ints
  int* tok_g = toks;
  int* tok_u = toks + Ecnt * MAXS;
  int* tok_d = toks + 2 * Ecnt * MAXS;
  float* probs  = (float*)(ws + 8192 + (size_t)3 * Ecnt * MAXS * sizeof(int));
  float* prob_g = probs;
  float* prob_u = probs + Ecnt * MAXS;
  float* prob_d = probs + 2 * Ecnt * MAXS;

  float* g = (float*)(ws + 2 * MB);             // [T,I] fp32 acc -> inter fp32
  float* u = (float*)(ws + 66 * MB);            // [T,I] fp32 acc -> wbf_down
  unsigned short* wgbf = (unsigned short*)(ws + 130 * MB);  // [E,I,H] bf16
  unsigned short* wubf = (unsigned short*)(ws + 194 * MB);  // [E,I,H] bf16
  unsigned short* xbf  = (unsigned short*)(ws + 258 * MB);  // [T,H]  bf16
  unsigned short* ibf  = (unsigned short*)(ws + 266 * MB);  // [T,I]  bf16
  unsigned short* wdbf = (unsigned short*)u;    // reuse u after SwiGLU (64MB)

  // -------- zero accumulation targets (graph-capturable) --------
  hipMemsetAsync(counts, 0, 3 * 8 * sizeof(int), stream);
  hipMemsetAsync(g, 0, 2 * (size_t)Tn * In * sizeof(float), stream);  // g + u
  hipMemsetAsync(out, 0, (size_t)Tn * Hn * sizeof(float), stream);

  dim3 blk(256);
  const long long nW = (long long)Ecnt * In * Hn;   // per weight tensor

  // bf16 pre-conversions
  cvt_bf16_kernel<<<2048, blk, 0, stream>>>(x, xbf, (long long)Tn * Hn);
  cvt_bf16_kernel<<<8192, blk, 0, stream>>>(w_gate, wgbf, nW);
  cvt_bf16_kernel<<<8192, blk, 0, stream>>>(w_up,   wubf, nW);

  // gate + up routing (fp32 activations)
  router_topk_kernel<<<Tn / 8, blk, 0, stream>>>(x, r_gate, Hn, counts_g, tok_g, prob_g);
  router_topk_kernel<<<Tn / 8, blk, 0, stream>>>(x, r_up,   Hn, counts_u, tok_u, prob_u);
  tilemap_kernel<<<1, 32, 0, stream>>>(counts_g, map_g);
  tilemap_kernel<<<1, 32, 0, stream>>>(counts_u, map_u);

  // gate + up routed GEMMs (H -> I)
  routed_gemm_wmma<<<dim3(MAX_TILES, In / 128), blk, 0, stream>>>(
      xbf, wgbf, g, map_g, counts_g, tok_g, prob_g, Hn, In);
  routed_gemm_wmma<<<dim3(MAX_TILES, In / 128), blk, 0, stream>>>(
      xbf, wubf, u, map_u, counts_u, tok_u, prob_u, Hn, In);

  // inter = silu(g) * u  -> g (fp32) + ibf (bf16)
  swiglu_kernel<<<4096, blk, 0, stream>>>(
      (float4*)g, (const float4*)u, (uint2*)ibf, Tn * In / 4);

  // w_down conversion into the (now free) u buffer
  cvt_bf16_kernel<<<8192, blk, 0, stream>>>(w_down, wdbf, nW);

  // down routing on fp32 inter + routed GEMM (I -> H) into d_out
  router_topk_kernel<<<Tn / 8, blk, 0, stream>>>(g, r_down, In, counts_d, tok_d, prob_d);
  tilemap_kernel<<<1, 32, 0, stream>>>(counts_d, map_d);
  routed_gemm_wmma<<<dim3(MAX_TILES, Hn / 128), blk, 0, stream>>>(
      ibf, wdbf, out, map_d, counts_d, tok_d, prob_d, In, Hn);
}
